// EABlock_43370579755390
// MI455X (gfx1250) — matrix operand order repaired and checked by reading, not compiled, not run on hardware
//
#include <hip/hip_runtime.h>

typedef __attribute__((ext_vector_type(16))) _Float16 v16h;
typedef __attribute__((ext_vector_type(8)))  _Float16 v8h;
typedef __attribute__((ext_vector_type(4)))  _Float16 v4h;
typedef __attribute__((ext_vector_type(8)))  float    v8f;

// ===================================================================
// setup / elementwise kernels
// ===================================================================

__global__ void bn_affine_k(const float* __restrict__ g, const float* __restrict__ b,
                            const float* __restrict__ m, const float* __restrict__ v,
                            float eps, float* __restrict__ scale, float* __restrict__ shift,
                            int C) {
  int c = blockIdx.x * blockDim.x + threadIdx.x;
  if (c < C) {
    float inv = g[c] * rsqrtf(v[c] + eps);
    scale[c] = inv;
    shift[c] = b[c] - m[c] * inv;
  }
}

// Fold per-input-channel bn affine into conv weight W[Cout][K], K = Cin*taps.
// Wh = f16(W * scale[ci]);  bias_out[co] = bias_in[co] + sum_k W[co][k]*shift[ci]
__global__ void fold_w_k(const float* __restrict__ W, const float* __restrict__ scale,
                         const float* __restrict__ shift, const float* __restrict__ bias_in,
                         _Float16* __restrict__ Wh, float* __restrict__ bias_out,
                         int K, int taps) {
  int co = blockIdx.x;
  const float* wrow = W + (long)co * K;
  _Float16* hrow = Wh + (long)co * K;
  __shared__ float red[256];
  float acc = 0.f;
  for (int k = threadIdx.x; k < K; k += 256) {
    int ci = k / taps;
    float w = wrow[k];
    hrow[k] = (_Float16)(w * scale[ci]);
    acc += w * shift[ci];
  }
  red[threadIdx.x] = acc;
  __syncthreads();
  for (int s = 128; s > 0; s >>= 1) {
    if (threadIdx.x < s) red[threadIdx.x] += red[threadIdx.x + s];
    __syncthreads();
  }
  if (threadIdx.x == 0 && bias_out)
    bias_out[co] = (bias_in ? bias_in[co] : 0.f) + red[0];
}

__global__ void convert_w_k(const float* __restrict__ W, _Float16* __restrict__ Wh, long n) {
  long i = (long)blockIdx.x * blockDim.x + threadIdx.x;
  if (i < n) Wh[i] = (_Float16)W[i];
}

__global__ void bn_act_k(const float* __restrict__ x, const float* __restrict__ scale,
                         const float* __restrict__ shift, float* __restrict__ y,
                         int C, long HW, long total, int relu) {
  long i = (long)blockIdx.x * blockDim.x + threadIdx.x;
  if (i < total) {
    int c = (int)((i / HW) % C);
    float v = x[i] * scale[c] + shift[c];
    if (relu) v = fmaxf(v, 0.f);
    y[i] = v;
  }
}

__global__ void bn_f16_k(const float* __restrict__ x, const float* __restrict__ scale,
                         const float* __restrict__ shift, _Float16* __restrict__ y,
                         int C, long HW, long total) {
  long i = (long)blockIdx.x * blockDim.x + threadIdx.x;
  if (i < total) {
    int c = (int)((i / HW) % C);
    y[i] = (_Float16)(x[i] * scale[c] + shift[c]);
  }
}

__global__ void gelu_k(float* __restrict__ x, long n) {
  long i = (long)blockIdx.x * blockDim.x + threadIdx.x;
  if (i < n) {
    float v = x[i];
    x[i] = 0.5f * v * (1.f + erff(v * 0.70710678118654752f));
  }
}

// _act_dn part 1: per (b,c) row of HW: softmax over spatial (in place)
__global__ void actdn_softmax_k(float* __restrict__ t, int HW) {
  float* x = t + (long)blockIdx.x * HW;
  __shared__ float red[256];
  int tid = threadIdx.x;
  float mx = -3.4e38f;
  for (int i = tid; i < HW; i += 256) mx = fmaxf(mx, x[i]);
  red[tid] = mx; __syncthreads();
  for (int s = 128; s > 0; s >>= 1) {
    if (tid < s) red[tid] = fmaxf(red[tid], red[tid + s]);
    __syncthreads();
  }
  mx = red[0];
  __syncthreads();
  float sum = 0.f;
  for (int i = tid; i < HW; i += 256) { float e = expf(x[i] - mx); x[i] = e; sum += e; }
  red[tid] = sum; __syncthreads();
  for (int s = 128; s > 0; s >>= 1) {
    if (tid < s) red[tid] += red[tid + s];
    __syncthreads();
  }
  float inv = 1.f / red[0];
  for (int i = tid; i < HW; i += 256) x[i] *= inv;
}

// _act_dn part 2: per (b,spatial): divide by (sum over 64 channels of head + 1e-6)
__global__ void actdn_div_k(float* __restrict__ t, int HW) {
  int idx = blockIdx.x * blockDim.x + threadIdx.x;
  if (idx >= 4 * HW) return;
  int b = idx / HW, s = idx % HW;
  for (int h = 0; h < 8; ++h) {
    float* base = t + ((long)(b * 512 + h * 64)) * HW + s;
    float sum = 0.f;
    for (int cc = 0; cc < 64; ++cc) sum += base[(long)cc * HW];
    float inv = 1.f / (sum + 1e-6f);
    for (int cc = 0; cc < 64; ++cc) base[(long)cc * HW] *= inv;
  }
}

// bn then 2x4 max-pool (24x48 -> 12x12); bn applied before max (scale may be <0)
__global__ void pool_bn_k(const float* __restrict__ xl, const float* __restrict__ sc,
                          const float* __restrict__ sh, float* __restrict__ out) {
  int idx = blockIdx.x * blockDim.x + threadIdx.x;
  if (idx >= 4 * 512 * 144) return;
  int b = idx / (512 * 144);
  int c = (idx / 144) % 512;
  int p = idx % 144;
  int i = p / 12, j = p % 12;
  const float* src = xl + ((long)(b * 512 + c)) * 1152;
  float a = sc[c], t = sh[c];
  float mx = -3.4e38f;
  for (int bi = 0; bi < 2; ++bi)
    for (int bj = 0; bj < 4; ++bj) {
      float v = src[(i * 2 + bi) * 48 + (j * 4 + bj)] * a + t;
      mx = fmaxf(mx, v);
    }
  out[idx] = mx;
}

// K3[b][pix][c] = ckvo[b][c][pix]  (ck half of ckv output), f16
__global__ void repack_k3_k(const float* __restrict__ ckvo, _Float16* __restrict__ K3) {
  int idx = blockIdx.x * blockDim.x + threadIdx.x;
  if (idx >= 4 * 144 * 128) return;
  int b = idx / (144 * 128);
  int p = (idx / 128) % 144;
  int c = idx % 128;
  K3[idx] = (_Float16)ckvo[((long)(b * 256 + c)) * 144 + p];
}

// V3[b][c][q] = ckvo[b][128+c][q] for q<144, else 0 (K padded 144->160)
__global__ void repack_v3_k(const float* __restrict__ ckvo, _Float16* __restrict__ V3) {
  int idx = blockIdx.x * blockDim.x + threadIdx.x;
  if (idx >= 4 * 128 * 160) return;
  int b = idx / (128 * 160);
  int c = (idx / 160) % 128;
  int q = idx % 160;
  V3[idx] = (q < 144) ? (_Float16)ckvo[((long)(b * 256 + 128 + c)) * 144 + q]
                      : (_Float16)0.f;
}

// out = x_h + bilinear_resize(cp, 24x48 -> 96x192), half-pixel centers, edge clamp
__global__ void add_resize_k(const float* __restrict__ xh, const float* __restrict__ cp,
                             float* __restrict__ out) {
  long idx = (long)blockIdx.x * blockDim.x + threadIdx.x;
  if (idx >= (long)4 * 128 * 96 * 192) return;
  int w = (int)(idx % 192);
  int h = (int)((idx / 192) % 96);
  long bc = idx / (192 * 96);
  float shf = (h + 0.5f) * 0.25f - 0.5f;
  float swf = (w + 0.5f) * 0.25f - 0.5f;
  float fh = shf - floorf(shf), fw = swf - floorf(swf);
  int h0 = (int)floorf(shf), w0 = (int)floorf(swf);
  int h0c = min(max(h0, 0), 23),  h1c = min(max(h0 + 1, 0), 23);
  int w0c = min(max(w0, 0), 47),  w1c = min(max(w0 + 1, 0), 47);
  const float* src = cp + bc * 1152;
  float v00 = src[h0c * 48 + w0c], v01 = src[h0c * 48 + w1c];
  float v10 = src[h1c * 48 + w0c], v11 = src[h1c * 48 + w1c];
  float bil = v00 * (1 - fh) * (1 - fw) + v01 * (1 - fh) * fw +
              v10 * fh * (1 - fw)       + v11 * fh * fw;
  out[idx] = xh[idx] + bil;
}

// attention column softmax over 144 keys (scale 1/12), write f16 P padded to 160 rows
__global__ void att_softmax_k(const float* __restrict__ S, _Float16* __restrict__ P, int N) {
  long idx = (long)blockIdx.x * blockDim.x + threadIdx.x;
  if (idx >= (long)4 * N) return;
  int b = (int)(idx / N), n = (int)(idx % N);
  const float* s = S + (long)b * 144 * N + n;
  _Float16* p = P + (long)b * 160 * N + n;
  const float sc = 1.f / 12.f;
  float mx = -3.4e38f;
  for (int k = 0; k < 144; ++k) mx = fmaxf(mx, s[(long)k * N] * sc);
  float sum = 0.f;
  for (int k = 0; k < 144; ++k) sum += expf(s[(long)k * N] * sc - mx);
  float inv = 1.f / sum;
  for (int k = 0; k < 144; ++k)
    p[(long)k * N] = (_Float16)(expf(s[(long)k * N] * sc - mx) * inv);
  for (int k = 144; k < 160; ++k) p[(long)k * N] = (_Float16)0.f;
}

// ===================================================================
// Generic implicit-GEMM WMMA kernel (wave32, v_wmma_f32_16x16x32_f16).
// MODE 0: B = f32 NCHW tensor, 1x1 conv (stride 1 or 2)
// MODE 1: B = f32 NCHW tensor, 3x3 conv pad=1 (stride 1 or 2), K = Cin*9
// MODE 2: B = f16 [K][N] matrix (per batch)
//
// Block = 128 threads = 4 waves; block tile = 64(M) x 32(N); K step 32.
// Double-buffered transposed LDS B-panel shBT[buf][n][k] (rows padded to 40
// halves -> conflict-free b128 fragment loads); branchless clamped staging
// (4 unconditional global loads -> one ds_store_b64 per group); A fragments
// as two 16B global loads per lane (k = 16*half + s: one contiguous
// 16-half run). One barrier per K step; next tile staged before the WMMAs.
// C[m][n] = sum_k A[m][k]*B[k][n] (+bias[m]) (+residual[same idx])
// ===================================================================
template<int MODE>
__global__ void gemm_wmma_k(const _Float16* __restrict__ A, long a_bstride,
                            const void* __restrict__ Bsrc, long b_bstride,
                            float* __restrict__ C, const float* __restrict__ bias,
                            const float* __restrict__ residual,
                            int M, int N, int K,
                            int Hin, int Win, int Wout, int stride) {
  __shared__ _Float16 shBT[2][32][40];
  const int tid  = threadIdx.x;
  const int wave = tid >> 5, lane = tid & 31;
  const int half = lane >> 4, r = lane & 15;
  const int n0 = blockIdx.x * 32;
  const int m0 = blockIdx.y * 64 + wave * 16;
  const int bb = blockIdx.z;
  const int mrow = m0 + r;
  const int mld  = (mrow < M) ? mrow : 0;
  const long HWin = (long)Hin * Win;

  const _Float16* Abase = A + (long)bb * a_bstride + (long)mld * K;

  // staging geometry: thread covers n = n0 + (tid&31), k = (tid>>5)*4 (+16)
  const int nn  = tid & 31;
  const int kqb = (tid >> 5) * 4;
  const int ng  = n0 + nn;
  const bool nok = ng < N;
  const int nc  = nok ? ng : 0;
  int oh = 0, ow = 0, pix = 0;
  if (MODE != 2) {
    oh = nc / Wout; ow = nc - oh * Wout;
    pix = (stride == 1) ? nc : (oh * stride * Win + ow * stride);
  }
  const float*    Xf = (MODE == 2) ? nullptr
                                   : (const float*)Bsrc + (long)bb * b_bstride;
  const _Float16* Xh = (MODE == 2) ? (const _Float16*)Bsrc + (long)bb * b_bstride
                                   : nullptr;

  auto stage = [&](int k0, int buf) {
    #pragma unroll
    for (int g = 0; g < 2; ++g) {
      const int kq = kqb + g * 16;
      v4h pk;
      #pragma unroll
      for (int i = 0; i < 4; ++i) {
        const int kg = k0 + kq + i;
        float val;
        if (MODE == 2) {
          val = (float)Xh[(long)kg * N + nc];
        } else if (MODE == 1) {
          int ci = kg / 9, tap = kg - ci * 9;
          int kh = tap / 3, kw = tap - kh * 3;
          int ih = oh * stride + kh - 1, iw = ow * stride + kw - 1;
          bool ok = (ih >= 0) & (ih < Hin) & (iw >= 0) & (iw < Win);
          long ad = ok ? ((long)ci * HWin + (long)ih * Win + iw) : 0;
          float v = Xf[ad];
          val = ok ? v : 0.f;
        } else {
          val = Xf[(long)kg * HWin + pix];
        }
        pk[i] = nok ? (_Float16)val : (_Float16)0.f;
      }
      *(v4h*)&shBT[buf][nn][kq] = pk;
      // prefetch the K+2 tile of the streamed operand (global_prefetch_b8)
      if (MODE == 0 && k0 + 32 < K)
        __builtin_prefetch(&Xf[(long)(k0 + 32 + kq) * HWin + pix], 0, 1);
      if (MODE == 2 && k0 + 32 < K)
        __builtin_prefetch(&Xh[(long)(k0 + 32 + kq) * N + nc], 0, 1);
    }
  };

  v8f acc0 = {}, acc1 = {};
  stage(0, 0);
  __syncthreads();
  int pb = 0;
  for (int k0 = 0; k0 < K; k0 += 32) {
    if (k0 + 32 < K) stage(k0 + 32, pb ^ 1);

    // A fragment: 16 contiguous halves per lane, k = 16*half + s (16B vector loads)
    v8h a0 = *(const v8h*)(Abase + k0 + 16 * half);
    v8h a1 = *(const v8h*)(Abase + k0 + 16 * half + 8);
    if (mrow >= M) { v8h z = {}; a0 = z; a1 = z; }
    v16h af;
    #pragma unroll
    for (int i = 0; i < 8; ++i) { af[i] = a0[i]; af[8 + i] = a1[i]; }

    // B fragments: 16 contiguous halves per lane from transposed LDS (b128 x2)
    const v8h* bp0 = (const v8h*)&shBT[pb][r][16 * half];
    const v8h* bp1 = (const v8h*)&shBT[pb][16 + r][16 * half];
    v8h b0a = bp0[0], b0b = bp0[1];
    v8h b1a = bp1[0], b1b = bp1[1];
    v16h bf0, bf1;
    #pragma unroll
    for (int i = 0; i < 8; ++i) {
      bf0[i] = b0a[i]; bf0[8 + i] = b0b[i];
      bf1[i] = b1a[i]; bf1[8 + i] = b1b[i];
    }

    acc0 = __builtin_amdgcn_wmma_f32_16x16x32_f16(false, af, false, bf0,
                                                  (short)0, acc0, false, false);
    acc1 = __builtin_amdgcn_wmma_f32_16x16x32_f16(false, af, false, bf1,
                                                  (short)0, acc1, false, false);
    __syncthreads();
    pb ^= 1;
  }

  long cbase = (long)bb * M * N;
  #pragma unroll
  for (int j = 0; j < 8; ++j) {
    int m = m0 + 8 * half + j;   // D layout: vgpr j -> M = j + 8*half, lane -> N
    if (m < M) {
      float bi = bias ? bias[m] : 0.f;
      long i1 = cbase + (long)m * N + n0 + r;       // n0+r < N (N mult of 16)
      float o1 = acc0[j] + bi;
      if (residual) o1 += residual[i1];
      C[i1] = o1;
      int c2 = n0 + 16 + r;
      if (c2 < N) {
        long i2 = i1 + 16;
        float o2 = acc1[j] + bi;
        if (residual) o2 += residual[i2];
        C[i2] = o2;
      }
    }
  }
}

// ===================================================================
// host side
// ===================================================================
static inline int cdiv(int a, int b) { return (a + b - 1) / b; }

struct Bump {
  char* base; size_t off;
  void* get(size_t bytes) {
    off = (off + 255) & ~(size_t)255;
    void* p = base + off;
    off += bytes;
    return p;
  }
};

extern "C" void kernel_launch(void* const* d_in, const int* in_sizes, int n_in,
                              void* d_out, int out_size, void* d_ws, size_t ws_size,
                              hipStream_t stream) {
  (void)in_sizes; (void)n_in; (void)out_size; (void)ws_size;
  auto F = [&](int i) { return (const float*)d_in[i]; };

  const float* x_h = F(0);
  const float* x_l = F(1);
  // param indices (dict order, bn tuples flattened g,b,m,v):
  // sc_bn 2-5, sc_w 6, al_bn 7-10, al_k 11, al_v 12, ml_bn 13-16, ml_w1 17,
  // ml_b1 18, ml_w2 19, ml_b2 20, cp_bn 21-24, cp_w 25, ckv_bn 26-29, ckv_w 30,
  // ah_bn 31-34, mh_bn 35-38, mh_w1 39, mh_b1 40, mh_w2 41, mh_b2 42,
  // dn_bn1 43-46, dn_w1 47, dn_bn2 48-51, dn_w2 52

  const int B = 4, Ch = 128, Cl_in = 256, Cl = 512;
  const int Hl = 48, Wl = 96, Hl2 = 24, Wl2 = 48;
  const int HWl2 = Hl2 * Wl2;                 // 1152
  const long HWh = 96L * 192;                 // 18432
  const long NXH = (long)B * Ch * HWh;        // 9437184

  Bump ws{(char*)d_ws, 0};
  auto affpair = [&](int C, float** s, float** t) {
    *s = (float*)ws.get((size_t)C * 4);
    *t = (float*)ws.get((size_t)C * 4);
  };
  float *sc_s, *sc_t, *al_s, *al_t, *ml_s, *ml_t, *cp_s, *cp_t, *ckv_s, *ckv_t;
  float *ah_s, *ah_t, *mh_s, *mh_t, *dn1_s, *dn1_t, *dn2_s, *dn2_t;
  affpair(Cl_in, &sc_s, &sc_t);   affpair(Cl_in, &al_s, &al_t);
  affpair(Cl, &ml_s, &ml_t);      affpair(Cl, &cp_s, &cp_t);
  affpair(Cl, &ckv_s, &ckv_t);    affpair(Ch, &ah_s, &ah_t);
  affpair(Ch, &mh_s, &mh_t);      affpair(Ch, &dn1_s, &dn1_t);
  affpair(Cl / 2, &dn2_s, &dn2_t);

  _Float16* Wsc  = (_Float16*)ws.get((size_t)Cl * Cl_in * 2);
  float*    bsc  = (float*)ws.get((size_t)Cl * 4);
  _Float16* Walk = (_Float16*)ws.get((size_t)Cl * Cl_in * 2);
  float*    balk = (float*)ws.get((size_t)Cl * 4);
  _Float16* Walv = (_Float16*)ws.get((size_t)Cl * Cl * 2);
  _Float16* Wml1 = (_Float16*)ws.get((size_t)Cl * Cl * 9 * 2);
  float*    bml1 = (float*)ws.get((size_t)Cl * 4);
  _Float16* Wml2 = (_Float16*)ws.get((size_t)Cl * Cl * 9 * 2);
  _Float16* Wcp  = (_Float16*)ws.get((size_t)Ch * Cl * 2);
  _Float16* Wckv = (_Float16*)ws.get((size_t)2 * Ch * Cl * 2);
  _Float16* Wmh1 = (_Float16*)ws.get((size_t)Ch * Ch * 9 * 2);
  float*    bmh1 = (float*)ws.get((size_t)Ch * 4);
  _Float16* Wmh2 = (_Float16*)ws.get((size_t)Ch * Ch * 9 * 2);
  _Float16* Wdn1 = (_Float16*)ws.get((size_t)(Cl / 2) * Ch * 9 * 2);
  _Float16* Wdn2 = (_Float16*)ws.get((size_t)Cl * (Cl / 2) * 9 * 2);

  float*    XL    = (float*)ws.get((size_t)B * Cl * HWl2 * 4);
  float*    M1    = (float*)ws.get((size_t)B * Cl * HWl2 * 4);
  float*    CPB   = (float*)ws.get((size_t)B * Ch * HWl2 * 4);
  float*    CPOOL = (float*)ws.get((size_t)B * Cl * 144 * 4);
  float*    CKVO  = (float*)ws.get((size_t)B * 2 * Ch * 144 * 4);
  _Float16* K3    = (_Float16*)ws.get((size_t)B * 144 * Ch * 2);
  _Float16* V3    = (_Float16*)ws.get((size_t)B * Ch * 160 * 2);
  _Float16* Tb    = (_Float16*)ws.get((size_t)NXH * 2);
  float*    Sb    = (float*)ws.get((size_t)B * 144 * HWh * 4);
  _Float16* Pb    = (_Float16*)ws.get((size_t)B * 160 * HWh * 2);
  float*    MH1   = (float*)ws.get((size_t)NXH * 4);
  float*    D1    = (float*)ws.get((size_t)B * (Cl / 2) * Hl * Wl * 4);

  float* xh_out = (float*)d_out;
  float* xl_out = (float*)d_out + NXH;

  // ---- bn affines ----
  auto bnaff = [&](int gi, float eps, float* s, float* t, int C) {
    bn_affine_k<<<cdiv(C, 256), 256, 0, stream>>>(F(gi), F(gi + 1), F(gi + 2), F(gi + 3),
                                                  eps, s, t, C);
  };
  bnaff(2, 1e-5f, sc_s, sc_t, Cl_in);
  bnaff(7, 1e-5f, al_s, al_t, Cl_in);
  bnaff(13, 1e-6f, ml_s, ml_t, Cl);
  bnaff(21, 1e-5f, cp_s, cp_t, Cl);
  bnaff(26, 1e-5f, ckv_s, ckv_t, Cl);
  bnaff(31, 1e-5f, ah_s, ah_t, Ch);
  bnaff(35, 1e-6f, mh_s, mh_t, Ch);
  bnaff(43, 1e-5f, dn1_s, dn1_t, Ch);
  bnaff(48, 1e-5f, dn2_s, dn2_t, Cl / 2);

  // ---- weight prep ----
  fold_w_k<<<Cl, 256, 0, stream>>>(F(6),  sc_s, sc_t, nullptr, Wsc,  bsc,  Cl_in, 1);
  fold_w_k<<<Cl, 256, 0, stream>>>(F(11), al_s, al_t, nullptr, Walk, balk, Cl_in, 1);
  convert_w_k<<<cdiv(Cl * Cl, 256), 256, 0, stream>>>(F(12), Walv, (long)Cl * Cl);
  fold_w_k<<<Cl, 256, 0, stream>>>(F(17), ml_s, ml_t, F(18), Wml1, bml1, Cl * 9, 9);
  convert_w_k<<<cdiv(Cl * Cl * 9, 256), 256, 0, stream>>>(F(19), Wml2, (long)Cl * Cl * 9);
  convert_w_k<<<cdiv(Ch * Cl, 256), 256, 0, stream>>>(F(25), Wcp, (long)Ch * Cl);
  convert_w_k<<<cdiv(2 * Ch * Cl, 256), 256, 0, stream>>>(F(30), Wckv, (long)2 * Ch * Cl);
  fold_w_k<<<Ch, 256, 0, stream>>>(F(39), mh_s, mh_t, F(40), Wmh1, bmh1, Ch * 9, 9);
  convert_w_k<<<cdiv(Ch * Ch * 9, 256), 256, 0, stream>>>(F(41), Wmh2, (long)Ch * Ch * 9);
  convert_w_k<<<cdiv((Cl / 2) * Ch * 9, 256), 256, 0, stream>>>(F(47), Wdn1, (long)(Cl / 2) * Ch * 9);
  convert_w_k<<<cdiv(Cl * (Cl / 2) * 9, 256), 256, 0, stream>>>(F(52), Wdn2, (long)Cl * (Cl / 2) * 9);

  const dim3 GB(128);
  // ---- low branch: x_l_res = conv1x1_s2(bn(x_l)) ----
  gemm_wmma_k<0><<<dim3(cdiv(HWl2, 32), cdiv(Cl, 64), B), GB, 0, stream>>>(
      Wsc, 0, x_l, (long)Cl_in * Hl * Wl, XL, bsc, nullptr,
      Cl, HWl2, Cl_in, Hl, Wl, Wl2, 2);
  // t = conv1x1_s2(bn(x_l))
  gemm_wmma_k<0><<<dim3(cdiv(HWl2, 32), cdiv(Cl, 64), B), GB, 0, stream>>>(
      Walk, 0, x_l, (long)Cl_in * Hl * Wl, M1, balk, nullptr,
      Cl, HWl2, Cl_in, Hl, Wl, Wl2, 2);
  // _act_dn
  actdn_softmax_k<<<B * Cl, 256, 0, stream>>>(M1, HWl2);
  actdn_div_k<<<cdiv(B * HWl2, 256), 256, 0, stream>>>(M1, HWl2);
  // x_l = x_l_res + conv1x1(t)
  gemm_wmma_k<0><<<dim3(cdiv(HWl2, 32), cdiv(Cl, 64), B), GB, 0, stream>>>(
      Walv, 0, M1, (long)Cl * HWl2, XL, nullptr, XL,
      Cl, HWl2, Cl, Hl2, Wl2, Wl2, 1);
  // ml: m = conv3x3(bn(x_l)) + b1
  gemm_wmma_k<1><<<dim3(cdiv(HWl2, 32), cdiv(Cl, 64), B), GB, 0, stream>>>(
      Wml1, 0, XL, (long)Cl * HWl2, M1, bml1, nullptr,
      Cl, HWl2, Cl * 9, Hl2, Wl2, Wl2, 1);
  gelu_k<<<cdiv((int)((long)B * Cl * HWl2), 256), 256, 0, stream>>>(M1, (long)B * Cl * HWl2);
  // x_l += conv3x3(gelu(m)) + b2
  gemm_wmma_k<1><<<dim3(cdiv(HWl2, 32), cdiv(Cl, 64), B), GB, 0, stream>>>(
      Wml2, 0, M1, (long)Cl * HWl2, XL, F(20), XL,
      Cl, HWl2, Cl * 9, Hl2, Wl2, Wl2, 1);

  // ---- cp: relu(bn(x_l)) -> 1x1 -> bilinear add into x_h ----
  bn_act_k<<<cdiv((int)((long)B * Cl * HWl2), 256), 256, 0, stream>>>(
      XL, cp_s, cp_t, M1, Cl, HWl2, (long)B * Cl * HWl2, 1);
  gemm_wmma_k<0><<<dim3(cdiv(HWl2, 32), cdiv(Ch, 64), B), GB, 0, stream>>>(
      Wcp, 0, M1, (long)Cl * HWl2, CPB, nullptr, nullptr,
      Ch, HWl2, Cl, Hl2, Wl2, Wl2, 1);
  add_resize_k<<<cdiv((int)NXH, 256), 256, 0, stream>>>(x_h, CPB, xh_out);

  // ---- cross K/V ----
  pool_bn_k<<<cdiv(B * Cl * 144, 256), 256, 0, stream>>>(XL, ckv_s, ckv_t, CPOOL);
  gemm_wmma_k<0><<<dim3(cdiv(144, 32), cdiv(2 * Ch, 64), B), GB, 0, stream>>>(
      Wckv, 0, CPOOL, (long)Cl * 144, CKVO, nullptr, nullptr,
      2 * Ch, 144, Cl, 12, 12, 12, 1);
  repack_k3_k<<<cdiv(B * 144 * Ch, 256), 256, 0, stream>>>(CKVO, K3);
  repack_v3_k<<<cdiv(B * Ch * 160, 256), 256, 0, stream>>>(CKVO, V3);

  // ---- cross attention on x_h ----
  bn_f16_k<<<cdiv((int)NXH, 256), 256, 0, stream>>>(xh_out, ah_s, ah_t, Tb, Ch, HWh, NXH);
  gemm_wmma_k<2><<<dim3(cdiv((int)HWh, 32), cdiv(144, 64), B), GB, 0, stream>>>(
      K3, (long)144 * Ch, Tb, (long)Ch * HWh, Sb, nullptr, nullptr,
      144, (int)HWh, Ch, 0, 0, 0, 1);
  att_softmax_k<<<cdiv((int)(B * HWh), 256), 256, 0, stream>>>(Sb, Pb, (int)HWh);
  gemm_wmma_k<2><<<dim3(cdiv((int)HWh, 32), cdiv(Ch, 64), B), GB, 0, stream>>>(
      V3, (long)Ch * 160, Pb, (long)160 * HWh, xh_out, nullptr, xh_out,
      Ch, (int)HWh, 160, 0, 0, 0, 1);

  // ---- mh mixer ----
  gemm_wmma_k<1><<<dim3(cdiv((int)HWh, 32), cdiv(Ch, 64), B), GB, 0, stream>>>(
      Wmh1, 0, xh_out, (long)Ch * HWh, MH1, bmh1, nullptr,
      Ch, (int)HWh, Ch * 9, 96, 192, 192, 1);
  gelu_k<<<cdiv((int)NXH, 256), 256, 0, stream>>>(MH1, NXH);
  gemm_wmma_k<1><<<dim3(cdiv((int)HWh, 32), cdiv(Ch, 64), B), GB, 0, stream>>>(
      Wmh2, 0, MH1, (long)Ch * HWh, xh_out, F(42), xh_out,
      Ch, (int)HWh, Ch * 9, 96, 192, 192, 1);

  // ---- down path ----
  bn_act_k<<<cdiv((int)NXH, 256), 256, 0, stream>>>(
      xh_out, dn1_s, dn1_t, MH1, Ch, HWh, NXH, 1);
  gemm_wmma_k<1><<<dim3(cdiv(Hl * Wl, 32), cdiv(Cl / 2, 64), B), GB, 0, stream>>>(
      Wdn1, 0, MH1, (long)Ch * HWh, D1, nullptr, nullptr,
      Cl / 2, Hl * Wl, Ch * 9, 96, 192, Wl, 2);
  bn_act_k<<<cdiv((int)((long)B * (Cl / 2) * Hl * Wl), 256), 256, 0, stream>>>(
      D1, dn2_s, dn2_t, D1, Cl / 2, (long)Hl * Wl, (long)B * (Cl / 2) * Hl * Wl, 1);
  gemm_wmma_k<1><<<dim3(cdiv(HWl2, 32), cdiv(Cl, 64), B), GB, 0, stream>>>(
      Wdn2, 0, D1, (long)(Cl / 2) * Hl * Wl, xl_out, nullptr, XL,
      Cl, HWl2, (Cl / 2) * 9, Hl, Wl, Wl2, 2);
}
